// RoPEAttention_13520557047920
// MI455X (gfx1250) — matrix operand order repaired and checked by reading, not compiled, not run on hardware
//
#include <hip/hip_runtime.h>
#include <hip/hip_bf16.h>

typedef __attribute__((ext_vector_type(16))) _Float16 v16h;
typedef __attribute__((ext_vector_type(8)))  _Float16 h8;
typedef __attribute__((ext_vector_type(8)))  float    v8f;
typedef __attribute__((ext_vector_type(4)))  unsigned int u32x4;
typedef __attribute__((ext_vector_type(4)))  int      i32x4;
typedef __attribute__((ext_vector_type(8)))  int      i32x8;

#define WMMA16(A, B, C) \
  __builtin_amdgcn_wmma_f32_16x16x32_f16(false, (A), false, (B), (short)0, (C), false, false)

// ---------------- Tensor Data Mover (TDM) tile load ----------------
// Builds a 2D D# (ISA ch.8): f16 elements (data_size=1 -> 2B), tile rows x width,
// row stride in elements, destination LDS byte offset. count=1, type=2 ("image").
// This toolchain's builtin is the 6-arg form:
//   (uint32x4 g0, int32x8 g1, int32x4 g2, int32x4 g3, int32x8 extra, i32 cpol)
#if defined(__gfx1250__) && __has_builtin(__builtin_amdgcn_tensor_load_to_lds)
#define HAVE_TDM 1
__device__ __forceinline__ void tdm_load_tile_2d(unsigned int lds_off, const void* gsrc,
                                                 unsigned int width_elems, unsigned int rows,
                                                 unsigned int stride_elems,
                                                 unsigned int tensor_rows) {
  const unsigned long long ga = (unsigned long long)(uintptr_t)gsrc;
  u32x4 g0;
  g0[0] = 1u;                                                   // count=1 (valid), user mode
  g0[1] = lds_off;                                              // lds_addr (bytes)
  g0[2] = (unsigned int)ga;                                     // global_addr[31:0]
  g0[3] = (unsigned int)((ga >> 32) & 0x01FFFFFFu) | (2u << 30); // addr[56:32] | type=2
  i32x8 g1;
  g1[0] = (int)(1u << 16);                                      // data_size=1 (2 bytes)
  g1[1] = (int)((width_elems & 0xFFFFu) << 16);                 // tensor_dim0[15:0]
  g1[2] = (int)(((width_elems >> 16) & 0xFFFFu) |
                ((tensor_rows & 0xFFFFu) << 16));               // dim0[31:16] | dim1[15:0]
  g1[3] = (int)(((tensor_rows >> 16) & 0xFFFFu) |
                ((width_elems & 0xFFFFu) << 16));               // dim1[31:16] | tile_dim0
  g1[4] = (int)(rows & 0xFFFFu);                                // tile_dim1 (tile_dim2=0)
  g1[5] = (int)stride_elems;                                    // tensor_dim0_stride[31:0]
  g1[6] = 0;                                                    // stride hi / dim1_stride lo
  g1[7] = 0;
  const i32x4 z4 = {0, 0, 0, 0};                                // groups 2/3 unused (2D)
  const i32x8 z8 = {0, 0, 0, 0, 0, 0, 0, 0};
  __builtin_amdgcn_tensor_load_to_lds(g0, g1, z4, z4, z8, 0);
}
#endif

// ---------- fragment loaders (contiguous k = 16*(lane/16) + s, wave32) ----------
__device__ __forceinline__ v16h ld_afrag(const _Float16* p, int lane) {
  const int b0 = (lane < 16) ? 0 : 16;
  h8 lo = *(const h8*)(p + b0);
  h8 hi = *(const h8*)(p + b0 + 8);
  v16h a;
#pragma unroll
  for (int i = 0; i < 8; ++i) { a[i] = lo[i]; a[i + 8] = hi[i]; }
  return a;
}

__device__ __forceinline__ v16h ld16(const _Float16* p) {
  h8 lo = *(const h8*)(p);
  h8 hi = *(const h8*)(p + 8);
  v16h b;
#pragma unroll
  for (int i = 0; i < 8; ++i) { b[i] = lo[i]; b[i + 8] = hi[i]; }
  return b;
}

// ---------- kernel 0: f32 -> f16 convert ----------
__global__ void __launch_bounds__(256) cvt_f32_f16(const float* __restrict__ in,
                                                   _Float16* __restrict__ out, int n) {
  for (int i = blockIdx.x * blockDim.x + threadIdx.x; i < n; i += gridDim.x * blockDim.x)
    out[i] = (_Float16)in[i];
}

// ---------- kernel 0b: RoPE cos/sin table [T=2048][D=64] ----------
__global__ void __launch_bounds__(256) rope_table(float* __restrict__ ct,
                                                  float* __restrict__ st) {
  int i = blockIdx.x * blockDim.x + threadIdx.x;  // 2048*64 threads
  int tok = i >> 6, d = i & 63;
  float theta = (float)tok * __powf(10000.0f, -(float)(d & 31) * (1.0f / 32.0f));
  float sn, cs;
  __sincosf(theta, &sn, &cs);
  ct[i] = cs;
  st[i] = sn;
}

// ---------- kernel 1: QKV GEMM (M=4096,N=3072,K=1024) + RoPE epilogue ----------
__global__ void __launch_bounds__(256) gemm_qkv_rope(const _Float16* __restrict__ xh,
                                                     const _Float16* __restrict__ wh,
                                                     const float* __restrict__ ct,
                                                     const float* __restrict__ st,
                                                     _Float16* __restrict__ qh,
                                                     _Float16* __restrict__ kh,
                                                     _Float16* __restrict__ vh) {
  const int lane = threadIdx.x & 31;
  const int wave = threadIdx.x >> 5;
  const int w    = blockIdx.x * 8 + wave;          // 12288 waves total
  const int mtile = w / 48;
  const int nch   = w % 48;
  const int nl    = lane & 15;
  const int klo   = (lane < 16) ? 0 : 16;

  const _Float16* arow = xh + (size_t)(mtile * 16 + nl) * 1024;
  const int n_ld = nch * 64 + nl;

  v8f c0 = {}, c1 = {}, c2 = {}, c3 = {};
  for (int k0 = 0; k0 < 1024; k0 += 32) {
    v16h a = ld_afrag(arow + k0, lane);
    c0 = WMMA16(a, ld16(wh + (size_t)(n_ld +  0) * 1024 + k0 + klo), c0);
    c1 = WMMA16(a, ld16(wh + (size_t)(n_ld + 16) * 1024 + k0 + klo), c1);
    c2 = WMMA16(a, ld16(wh + (size_t)(n_ld + 32) * 1024 + k0 + klo), c2);
    c3 = WMMA16(a, ld16(wh + (size_t)(n_ld + 48) * 1024 + k0 + klo), c3);
  }

  const int half8 = (lane < 16) ? 0 : 8;
#pragma unroll
  for (int r = 0; r < 8; ++r) {
    const int m   = mtile * 16 + r + half8;
    const int bb  = m >> 11;
    const int tok = m & 2047;
    float av[4] = { c0[r], c1[r], c2[r], c3[r] };
#pragma unroll
    for (int t = 0; t < 4; ++t) {
      const int n    = nch * 64 + t * 16 + nl;
      const int sel  = n >> 10;           // 0=q 1=k 2=v
      const int cc   = n & 1023;
      const int head = cc >> 6;
      const int d    = cc & 63;
      float v = av[t];
      float partner = __shfl_xor(v, 1, 32);   // x[d^1] (same row, adjacent feature)
      if (sel < 2) {
        const float cs = ct[tok * 64 + d];
        const float sn = st[tok * 64 + d];
        v = (d & 1) ? (v * cs + partner * sn) : (v * cs - partner * sn);
      }
      const size_t oidx = (((size_t)bb * 16 + head) * 2048 + tok) * 64 + d;
      if (sel == 0)      qh[oidx] = (_Float16)(v * 0.125f);  // fold 1/sqrt(64)
      else if (sel == 1) kh[oidx] = (_Float16)v;
      else               vh[oidx] = (_Float16)v;
    }
  }
}

// ---------- kernel 2: flash attention (TDM-staged K/V tiles) ----------
__global__ void __launch_bounds__(256) flash_attn(const _Float16* __restrict__ qh,
                                                  const _Float16* __restrict__ kh,
                                                  const _Float16* __restrict__ vh,
                                                  _Float16* __restrict__ yh) {
  __shared__ __align__(32) _Float16 Ks[32][64];      // K tile [n][d]
  __shared__ __align__(32) _Float16 Vs[32][64];      // V tile [n][d] (TDM dest)
  __shared__ __align__(32) _Float16 Vt[64][32];      // V tile transposed [d][n]
  __shared__ __align__(32) _Float16 Ps[8][16][32];   // per-wave P scratch

  const int lane = threadIdx.x & 31;
  const int wave = threadIdx.x >> 5;
  const int bh   = blockIdx.x >> 4;
  const int qch  = blockIdx.x & 15;
  const int b    = bh >> 4;
  const int h    = bh & 15;
  const int nl   = lane & 15;
  const int klo  = (lane < 16) ? 0 : 16;
  const int half8 = (lane < 16) ? 0 : 8;

  const _Float16* Qb = qh + (size_t)bh * 2048 * 64;
  const _Float16* Kb = kh + (size_t)bh * 2048 * 64;
  const _Float16* Vb = vh + (size_t)bh * 2048 * 64;

  const int q0 = qch * 128 + wave * 16;
  const _Float16* qrow = Qb + (size_t)(q0 + nl) * 64;
  v16h qa0 = ld_afrag(qrow,      lane);
  v16h qa1 = ld_afrag(qrow + 32, lane);

  v8f o0 = {}, o1 = {}, o2 = {}, o3 = {};
  float mrow[8], lrow[8];
#pragma unroll
  for (int r = 0; r < 8; ++r) { mrow[r] = -1e30f; lrow[r] = 0.0f; }

  const int ldrow = threadIdx.x >> 3;
  const int ldc8  = (threadIdx.x & 7) << 3;

  for (int kv0 = 0; kv0 < 2048; kv0 += 32) {
    __syncthreads();     // previous tile fully consumed
#if defined(HAVE_TDM)
    if (wave == 0) {     // one TDM descriptor each for K and V tiles (EXEC ignored)
      tdm_load_tile_2d((unsigned int)(uintptr_t)&Ks[0][0],
                       Kb + (size_t)kv0 * 64, 64u, 32u, 64u, 2048u);
      tdm_load_tile_2d((unsigned int)(uintptr_t)&Vs[0][0],
                       Vb + (size_t)kv0 * 64, 64u, 32u, 64u, 2048u);
      __builtin_amdgcn_s_wait_tensorcnt((short)0);
    }
    __syncthreads();     // LDS tiles visible to all waves
    {                    // transpose V tile in LDS: Vs[n][d] -> Vt[d][n]
      h8 vv = *(const h8*)(&Vs[ldrow][ldc8]);
#pragma unroll
      for (int i = 0; i < 8; ++i) Vt[ldc8 + i][ldrow] = vv[i];
    }
#else
    {
      h8 kk = *(const h8*)(Kb + (size_t)(kv0 + ldrow) * 64 + ldc8);
      *(h8*)(&Ks[ldrow][ldc8]) = kk;
      h8 vv = *(const h8*)(Vb + (size_t)(kv0 + ldrow) * 64 + ldc8);
#pragma unroll
      for (int i = 0; i < 8; ++i) Vt[ldc8 + i][ldrow] = vv[i];
      if (kv0 + 32 < 2048) {
        __builtin_prefetch(Kb + (size_t)(kv0 + 32 + ldrow) * 64 + ldc8, 0, 0);
        __builtin_prefetch(Vb + (size_t)(kv0 + 32 + ldrow) * 64 + ldc8, 0, 0);
      }
    }
#endif
    __syncthreads();

    // S = Q * K^T  (scale pre-folded into Q)
    v8f sa = {}, sb = {};
    sa = WMMA16(qa0, ld16(&Ks[nl][klo]),           sa);
    sa = WMMA16(qa1, ld16(&Ks[nl][klo + 32]),      sa);
    sb = WMMA16(qa0, ld16(&Ks[16 + nl][klo]),      sb);
    sb = WMMA16(qa1, ld16(&Ks[16 + nl][klo + 32]), sb);

    // online softmax
#pragma unroll
    for (int r = 0; r < 8; ++r) {
      float va = sa[r], vb = sb[r];
      float mx = fmaxf(va, vb);
      mx = fmaxf(mx, __shfl_xor(mx, 1, 16));
      mx = fmaxf(mx, __shfl_xor(mx, 2, 16));
      mx = fmaxf(mx, __shfl_xor(mx, 4, 16));
      mx = fmaxf(mx, __shfl_xor(mx, 8, 16));
      float mnew = fmaxf(mrow[r], mx);
      float corr = __expf(mrow[r] - mnew);
      mrow[r] = mnew;
      float pa = __expf(va - mnew);
      float pb = __expf(vb - mnew);
      float rs = pa + pb;
      rs += __shfl_xor(rs, 1, 16);
      rs += __shfl_xor(rs, 2, 16);
      rs += __shfl_xor(rs, 4, 16);
      rs += __shfl_xor(rs, 8, 16);
      lrow[r] = lrow[r] * corr + rs;
      o0[r] *= corr; o1[r] *= corr; o2[r] *= corr; o3[r] *= corr;
      const int M = r + half8;
      Ps[wave][M][nl]      = (_Float16)pa;
      Ps[wave][M][16 + nl] = (_Float16)pb;
    }

    // O += P * V
    v16h pA = ld_afrag(&Ps[wave][nl][0], lane);
    o0 = WMMA16(pA, ld16(&Vt[ 0 + nl][klo]), o0);
    o1 = WMMA16(pA, ld16(&Vt[16 + nl][klo]), o1);
    o2 = WMMA16(pA, ld16(&Vt[32 + nl][klo]), o2);
    o3 = WMMA16(pA, ld16(&Vt[48 + nl][klo]), o3);
  }

#pragma unroll
  for (int r = 0; r < 8; ++r) {
    const int tok = q0 + r + half8;
    const float inv = 1.0f / lrow[r];
    const size_t base = ((size_t)b * 2048 + tok) * 1024 + h * 64;
    yh[base +  0 + nl] = (_Float16)(o0[r] * inv);
    yh[base + 16 + nl] = (_Float16)(o1[r] * inv);
    yh[base + 32 + nl] = (_Float16)(o2[r] * inv);
    yh[base + 48 + nl] = (_Float16)(o3[r] * inv);
  }
}

// ---------- kernel 3: output GEMM (M=4096,N=1024,K=1024), f32 out ----------
__global__ void __launch_bounds__(256) gemm_out(const _Float16* __restrict__ yh,
                                                const _Float16* __restrict__ wh,
                                                float* __restrict__ out) {
  const int lane = threadIdx.x & 31;
  const int wave = threadIdx.x >> 5;
  const int w    = blockIdx.x * 8 + wave;
  const int mtile = w / 16;
  const int nch   = w % 16;
  const int nl    = lane & 15;
  const int klo   = (lane < 16) ? 0 : 16;

  const _Float16* arow = yh + (size_t)(mtile * 16 + nl) * 1024;
  const int n_ld = nch * 64 + nl;

  v8f c0 = {}, c1 = {}, c2 = {}, c3 = {};
  for (int k0 = 0; k0 < 1024; k0 += 32) {
    v16h a = ld_afrag(arow + k0, lane);
    c0 = WMMA16(a, ld16(wh + (size_t)(n_ld +  0) * 1024 + k0 + klo), c0);
    c1 = WMMA16(a, ld16(wh + (size_t)(n_ld + 16) * 1024 + k0 + klo), c1);
    c2 = WMMA16(a, ld16(wh + (size_t)(n_ld + 32) * 1024 + k0 + klo), c2);
    c3 = WMMA16(a, ld16(wh + (size_t)(n_ld + 48) * 1024 + k0 + klo), c3);
  }

  const int half8 = (lane < 16) ? 0 : 8;
#pragma unroll
  for (int r = 0; r < 8; ++r) {
    const size_t m = (size_t)(mtile * 16 + r + half8);
    out[m * 1024 + nch * 64 +  0 + nl] = c0[r];
    out[m * 1024 + nch * 64 + 16 + nl] = c1[r];
    out[m * 1024 + nch * 64 + 32 + nl] = c2[r];
    out[m * 1024 + nch * 64 + 48 + nl] = c3[r];
  }
}

extern "C" void kernel_launch(void* const* d_in, const int* in_sizes, int n_in,
                              void* d_out, int out_size, void* d_ws, size_t ws_size,
                              hipStream_t stream) {
  const float* x    = (const float*)d_in[0];   // [2,2048,1024]
  const float* Wqkv = (const float*)d_in[1];   // [3072,1024]
  const float* Wout = (const float*)d_in[2];   // [1024,1024]
  float* out = (float*)d_out;                  // [2,2048,1024]

  char* ws = (char*)d_ws;
  _Float16* xh    = (_Float16*)(ws + 0);          // 8 MB
  _Float16* wqkvh = (_Float16*)(ws + 8388608);    // 6 MB
  _Float16* wouth = (_Float16*)(ws + 14680064);   // 2 MB
  _Float16* qh    = (_Float16*)(ws + 16777216);   // 8 MB [B,H,T,D]
  _Float16* kh    = (_Float16*)(ws + 25165824);   // 8 MB
  _Float16* vh    = (_Float16*)(ws + 33554432);   // 8 MB
  _Float16* yh    = (_Float16*)(ws + 41943040);   // 8 MB [B,T,C]
  float*    ct    = (float*)   (ws + 50331648);   // 512 KB cos table
  float*    st    = (float*)   (ws + 50855936);   // 512 KB sin table

  cvt_f32_f16<<<4096, 256, 0, stream>>>(x,    xh,    2 * 2048 * 1024);
  cvt_f32_f16<<<3072, 256, 0, stream>>>(Wqkv, wqkvh, 3072 * 1024);
  cvt_f32_f16<<<1024, 256, 0, stream>>>(Wout, wouth, 1024 * 1024);
  rope_table<<<512, 256, 0, stream>>>(ct, st);    // 2048*64 elems

  gemm_qkv_rope<<<1536, 256, 0, stream>>>(xh, wqkvh, ct, st, qh, kh, vh);
  flash_attn<<<512, 256, 0, stream>>>(qh, kh, vh, yh);
  gemm_out<<<512, 256, 0, stream>>>(yh, wouth, out);
}
